// SimulatorModel_67886253080809
// MI455X (gfx1250) — compile-verified, matrix operand order checked by reading, not run on hardware
//
#include <hip/hip_runtime.h>
#include <hip/hip_bf16.h>
#include <math.h>

typedef __attribute__((ext_vector_type(16))) _Float16 v16h;
typedef __attribute__((ext_vector_type(8)))  float    v8f;

#define NN 50000
#define EE 800000
#define HD 64
#define SN 40   // padded LDS stride for 16x32 input tiles (halfs)
#define SH 72   // padded LDS stride for 16x64 hidden tiles (halfs)
#define NFRAG 50
#define TQ_E 10  // tiles per wave, edge kernel   (50000 = 625 blk * 8 w * 10)
#define TQ_N 5   // tiles per wave, node/decoder  (3125  = 125 blk * 5 w * 5)

// K-index striping for 16-bit A/B WMMA fragments (ISA 7.12.2):
// lane group g = lane/16 covers K sub-blocks {g*8..g*8+7, 16+g*8..16+g*8+7}
__device__ __forceinline__ int kpat(int e, int g) {
  return ((e >> 3) << 4) + (g << 3) + (e & 7);
}

// Build a 16x32 f16 A fragment from an LDS tile buf[m*stride + k]
__device__ __forceinline__ v16h load_a(const _Float16* buf, int stride, int lane) {
  const int g = lane >> 4, m = lane & 15;
  v16h a;
#pragma unroll
  for (int e = 0; e < 16; ++e) a[e] = buf[m * stride + kpat(e, g)];
  return a;
}

__device__ __forceinline__ v8f wmma16(v16h a, v16h b, v8f c) {
  return __builtin_amdgcn_wmma_f32_16x16x32_f16(false, a, false, b, (short)0, c,
                                                false, false);
}

// ---------------------------------------------------------------------------
// Pack all weight matrices into fragment-major f16:  wf[frag][lane][16 halfs]
// frag map:
//  edge : 0-3 (ew1, ntile), 4-11 (ew2, ntile*2+kb), 12-13 (ew3, kb)
//  node : 14-17, 18-25, 26-27
//  dec  : 28-31 (dw1), 32-39 (dw2), 40-47 (dw3), 48-49 (dw4)
// ---------------------------------------------------------------------------
__global__ void k_pack(const float* ew1, const float* ew2, const float* ew3,
                       const float* nw1, const float* nw2, const float* nw3,
                       const float* dw1, const float* dw2, const float* dw3,
                       const float* dw4, _Float16* wf) {
  int tid = blockIdx.x * blockDim.x + threadIdx.x;
  if (tid >= NFRAG * 32) return;
  int f = tid >> 5, lane = tid & 31;
  const float* W = ew1;
  int Kdim = 9, Ndim = 64, nt = 0, kb = 0;
  int fl = f;
  if (fl < 14) {
    if (fl < 4)       { W = ew1; Kdim = 9;  Ndim = 64; nt = fl; }
    else if (fl < 12) { W = ew2; Kdim = 64; Ndim = 64; nt = (fl - 4) >> 1; kb = (fl - 4) & 1; }
    else              { W = ew3; Kdim = 64; Ndim = 3;  kb = fl - 12; }
  } else if (fl < 28) {
    fl -= 14;
    if (fl < 4)       { W = nw1; Kdim = 5;  Ndim = 64; nt = fl; }
    else if (fl < 12) { W = nw2; Kdim = 64; Ndim = 64; nt = (fl - 4) >> 1; kb = (fl - 4) & 1; }
    else              { W = nw3; Kdim = 64; Ndim = 1;  kb = fl - 12; }
  } else {
    fl -= 28;
    if (fl < 4)       { W = dw1; Kdim = 5;  Ndim = 64; nt = fl; }
    else if (fl < 12) { W = dw2; Kdim = 64; Ndim = 64; nt = (fl - 4) >> 1; kb = (fl - 4) & 1; }
    else if (fl < 20) { W = dw3; Kdim = 64; Ndim = 64; nt = (fl - 12) >> 1; kb = (fl - 12) & 1; }
    else              { W = dw4; Kdim = 64; Ndim = 1;  kb = fl - 20; }
  }
  int g = lane >> 4, nloc = lane & 15;
  int n = nt * 16 + nloc;
  _Float16* out = wf + (size_t)(f * 32 + lane) * 16;
#pragma unroll
  for (int e = 0; e < 16; ++e) {
    int k = kb * 32 + kpat(e, g);
    float v = (k < Kdim && n < Ndim) ? W[k * Ndim + n] : 0.0f;
    out[e] = (_Float16)v;
  }
}

// ---------------------------------------------------------------------------
__global__ void k_node_init(const float* X, const float* yp, float* x_in,
                            float* cnt) {
  int n = blockIdx.x * blockDim.x + threadIdx.x;
  if (n >= NN) return;
  x_in[n * 5 + 0] = X[n * 6 + 0];
  x_in[n * 5 + 1] = X[n * 6 + 1];
  x_in[n * 5 + 2] = X[n * 6 + 2];
  x_in[n * 5 + 3] = X[n * 6 + 4];
  x_in[n * 5 + 4] = yp[n];
  cnt[n] = 0.0f;
}

__global__ void k_edge_init(const int* edge, const float* x_in, float* ea,
                            float* cnt) {
  int e = blockIdx.x * blockDim.x + threadIdx.x;
  if (e >= EE) return;
  int s = edge[e], d = edge[EE + e];
  float ux = x_in[d * 5 + 0] - x_in[s * 5 + 0];
  float uy = x_in[d * 5 + 1] - x_in[s * 5 + 1];
  float uz = x_in[d * 5 + 2] - x_in[s * 5 + 2];
  float fr = x_in[d * 5 + 4] - x_in[s * 5 + 4];
  ea[e * 3 + 0] = fr * ux;
  ea[e * 3 + 1] = fr * uy;
  ea[e * 3 + 2] = fr * uz;
  atomicAdd(&cnt[d], 1.0f);
}

__global__ void k_zero(float* p, int n) {
  int i = blockIdx.x * blockDim.x + threadIdx.x;
  if (i < n) p[i] = 0.0f;
}

// ---------------------------------------------------------------------------
// Edge MLP step: one wave processes TQ_E 16-edge tiles with all 14 weight
// B-fragments register-resident (112 VGPRs). 14 WMMAs per tile.
// Fuses: feature build, 9->64->64->3 MLP, edge_attr update, scatter-sum.
// ---------------------------------------------------------------------------
__global__ __launch_bounds__(256) void k_edge_mlp(
    const int* __restrict__ edge, const float* __restrict__ x_in,
    float* __restrict__ ea, float* __restrict__ aggr,
    const _Float16* __restrict__ wfg, const float* __restrict__ eb1,
    const float* __restrict__ eb2, const float* __restrict__ eb3) {
  __shared__ _Float16 sNet[8][16 * SN];
  __shared__ _Float16 sH[8][16 * SH];
  const int lane = threadIdx.x & 31;
  const int w = threadIdx.x >> 5;
  const int wave = blockIdx.x * 8 + w;
  const int g = lane >> 4, nloc = lane & 15, m = lane & 15;
  const v16h* wfv = (const v16h*)wfg;

  // hoist all weight fragments into registers (shared by every tile)
  v16h bf[14];
#pragma unroll
  for (int i = 0; i < 14; ++i) bf[i] = wfv[i * 32 + lane];

  // biases are also tile-invariant
  float b1a[4], b2a[4];
#pragma unroll
  for (int nt = 0; nt < 4; ++nt) {
    b1a[nt] = eb1[nt * 16 + nloc];
    b2a[nt] = eb2[nt * 16 + nloc];
  }
  float b3 = eb3[nloc < 3 ? nloc : 0];

  // one-time zero of the padding columns of the input staging tile
  if (lane < 16) {
#pragma unroll
    for (int k = 9; k < 32; ++k) sNet[w][m * SN + k] = (_Float16)0.0f;
  }
  __syncthreads();

  const v8f cz = {};
  for (int it = 0; it < TQ_E; ++it) {
    const int tile = wave * TQ_E + it;

    // phase 1: per-edge features; both lane halves work on edge m
    {
      int e = tile * 16 + m;
      int s = edge[e], d = edge[EE + e];
      _Float16* row = &sNet[w][m * SN];
      if (g == 0) {
        float ux = x_in[d * 5 + 0] - x_in[s * 5 + 0];
        float uy = x_in[d * 5 + 1] - x_in[s * 5 + 1];
        float uz = x_in[d * 5 + 2] - x_in[s * 5 + 2];
        float nrm = sqrtf(ux * ux + uy * uy + uz * uz + 1e-12f);
        row[0] = (_Float16)ux;
        row[1] = (_Float16)uy;
        row[2] = (_Float16)uz;
        row[3] = (_Float16)nrm;
      } else {
        row[4] = (_Float16)ea[e * 3 + 0];
        row[5] = (_Float16)ea[e * 3 + 1];
        row[6] = (_Float16)ea[e * 3 + 2];
        row[7] = (_Float16)x_in[s * 5 + 4];
        row[8] = (_Float16)x_in[d * 5 + 4];
      }
    }
    __syncthreads();

    v16h a1 = load_a(&sNet[w][0], SN, lane);
    v8f acc[4];
#pragma unroll
    for (int nt = 0; nt < 4; ++nt) acc[nt] = wmma16(a1, bf[nt], cz);
#pragma unroll
    for (int nt = 0; nt < 4; ++nt) {
#pragma unroll
      for (int r = 0; r < 8; ++r) {
        float h = acc[nt][r] + b1a[nt];
        sH[w][(r + g * 8) * SH + nt * 16 + nloc] = (_Float16)(h > 0.f ? h : 0.f);
      }
    }
    __syncthreads();

    v16h a20 = load_a(&sH[w][0], SH, lane);
    v16h a21 = load_a(&sH[w][32], SH, lane);
#pragma unroll
    for (int nt = 0; nt < 4; ++nt) {
      v8f c = wmma16(a20, bf[4 + nt * 2 + 0], cz);
      acc[nt] = wmma16(a21, bf[4 + nt * 2 + 1], c);
    }
    __syncthreads();
#pragma unroll
    for (int nt = 0; nt < 4; ++nt) {
#pragma unroll
      for (int r = 0; r < 8; ++r) {
        float h = acc[nt][r] + b2a[nt];
        sH[w][(r + g * 8) * SH + nt * 16 + nloc] = (_Float16)(h > 0.f ? h : 0.f);
      }
    }
    __syncthreads();

    v16h a30 = load_a(&sH[w][0], SH, lane);
    v16h a31 = load_a(&sH[w][32], SH, lane);
    v8f c3 = wmma16(a30, bf[12], cz);
    c3 = wmma16(a31, bf[13], c3);

    if (nloc < 3) {
#pragma unroll
      for (int r = 0; r < 8; ++r) {
        int eidx = tile * 16 + r + g * 8;
        float v = ea[eidx * 3 + nloc] + c3[r] + b3;  // residual update
        ea[eidx * 3 + nloc] = v;
        int d = edge[EE + eidx];
        atomicAdd(&aggr[d * 3 + nloc], v);           // scatter-sum
      }
    }
    __syncthreads();  // sH/sNet reuse boundary before next tile
  }
}

// ---------------------------------------------------------------------------
// Node MLP step: one wave processes TQ_N 16-node tiles; 14 WMMAs per tile,
// weight fragments register-resident.
// ---------------------------------------------------------------------------
__global__ __launch_bounds__(160) void k_node_mlp(
    float* __restrict__ x_in, const float* __restrict__ aggr,
    const float* __restrict__ cnt, const _Float16* __restrict__ wfg,
    const float* __restrict__ nb1, const float* __restrict__ nb2,
    const float* __restrict__ nb3) {
  __shared__ _Float16 sNet[5][16 * SN];
  __shared__ _Float16 sH[5][16 * SH];
  const int lane = threadIdx.x & 31;
  const int w = threadIdx.x >> 5;
  const int wave = blockIdx.x * 5 + w;
  const int g = lane >> 4, nloc = lane & 15, m = lane & 15;
  const v16h* wfv = (const v16h*)wfg;

  v16h bf[14];
#pragma unroll
  for (int i = 0; i < 14; ++i) bf[i] = wfv[(14 + i) * 32 + lane];

  float b1a[4], b2a[4];
#pragma unroll
  for (int nt = 0; nt < 4; ++nt) {
    b1a[nt] = nb1[nt * 16 + nloc];
    b2a[nt] = nb2[nt * 16 + nloc];
  }
  float b3 = nb3[0];

  if (lane < 16) {
#pragma unroll
    for (int k = 5; k < 32; ++k) sNet[w][m * SN + k] = (_Float16)0.0f;
  }
  __syncthreads();

  const v8f cz = {};
  for (int it = 0; it < TQ_N; ++it) {
    const int tile = wave * TQ_N + it;

    if (lane < 16) {
      int n = tile * 16 + m;
      float dn = cnt[n];
      dn = dn > 1.f ? dn : 1.f;
      _Float16* row = &sNet[w][m * SN];
      row[0] = (_Float16)x_in[n * 5 + 3];
      row[1] = (_Float16)x_in[n * 5 + 4];
      row[2] = (_Float16)(aggr[n * 3 + 0] / dn);
      row[3] = (_Float16)(aggr[n * 3 + 1] / dn);
      row[4] = (_Float16)(aggr[n * 3 + 2] / dn);
    }
    __syncthreads();

    v16h a1 = load_a(&sNet[w][0], SN, lane);
    v8f acc[4];
#pragma unroll
    for (int nt = 0; nt < 4; ++nt) acc[nt] = wmma16(a1, bf[nt], cz);
#pragma unroll
    for (int nt = 0; nt < 4; ++nt) {
#pragma unroll
      for (int r = 0; r < 8; ++r) {
        float h = acc[nt][r] + b1a[nt];
        sH[w][(r + g * 8) * SH + nt * 16 + nloc] = (_Float16)(h > 0.f ? h : 0.f);
      }
    }
    __syncthreads();

    v16h a20 = load_a(&sH[w][0], SH, lane);
    v16h a21 = load_a(&sH[w][32], SH, lane);
#pragma unroll
    for (int nt = 0; nt < 4; ++nt) {
      v8f c = wmma16(a20, bf[4 + nt * 2 + 0], cz);
      acc[nt] = wmma16(a21, bf[4 + nt * 2 + 1], c);
    }
    __syncthreads();
#pragma unroll
    for (int nt = 0; nt < 4; ++nt) {
#pragma unroll
      for (int r = 0; r < 8; ++r) {
        float h = acc[nt][r] + b2a[nt];
        sH[w][(r + g * 8) * SH + nt * 16 + nloc] = (_Float16)(h > 0.f ? h : 0.f);
      }
    }
    __syncthreads();

    v16h a30 = load_a(&sH[w][0], SH, lane);
    v16h a31 = load_a(&sH[w][32], SH, lane);
    v8f c3 = wmma16(a30, bf[12], cz);
    c3 = wmma16(a31, bf[13], c3);

    if (nloc == 0) {
#pragma unroll
      for (int r = 0; r < 8; ++r) {
        int n = tile * 16 + r + g * 8;
        x_in[n * 5 + 4] += c3[r] + b3;   // residual scalar-field update
      }
    }
    __syncthreads();
  }
}

// ---------------------------------------------------------------------------
// Decoder: 5->64->64->64->1; one wave per TQ_N 16-node tiles; 22 WMMAs/tile,
// weight fragments register-resident (176 VGPRs).
// ---------------------------------------------------------------------------
__global__ __launch_bounds__(160) void k_decoder(
    const float* __restrict__ x_in, const float* __restrict__ yp,
    const _Float16* __restrict__ wfg, const float* __restrict__ db1,
    const float* __restrict__ db2, const float* __restrict__ db3,
    const float* __restrict__ db4, float* __restrict__ out) {
  __shared__ _Float16 sNet[5][16 * SN];
  __shared__ _Float16 sH[5][16 * SH];
  const int lane = threadIdx.x & 31;
  const int w = threadIdx.x >> 5;
  const int wave = blockIdx.x * 5 + w;
  const int g = lane >> 4, nloc = lane & 15, m = lane & 15;
  const v16h* wfv = (const v16h*)wfg;

  v16h bf[22];
#pragma unroll
  for (int i = 0; i < 22; ++i) bf[i] = wfv[(28 + i) * 32 + lane];

  float b1a[4], b2a[4], b3a[4];
#pragma unroll
  for (int nt = 0; nt < 4; ++nt) {
    b1a[nt] = db1[nt * 16 + nloc];
    b2a[nt] = db2[nt * 16 + nloc];
    b3a[nt] = db3[nt * 16 + nloc];
  }
  float b4 = db4[0];

  if (lane < 16) {
#pragma unroll
    for (int k = 5; k < 32; ++k) sNet[w][m * SN + k] = (_Float16)0.0f;
  }
  __syncthreads();

  const v8f cz = {};
  for (int it = 0; it < TQ_N; ++it) {
    const int tile = wave * TQ_N + it;

    if (lane < 16) {
      int n = tile * 16 + m;
      _Float16* row = &sNet[w][m * SN];
#pragma unroll
      for (int k = 0; k < 5; ++k) row[k] = (_Float16)x_in[n * 5 + k];
    }
    __syncthreads();

    v16h a1 = load_a(&sNet[w][0], SN, lane);
    v8f acc[4];
#pragma unroll
    for (int nt = 0; nt < 4; ++nt) acc[nt] = wmma16(a1, bf[nt], cz);
#pragma unroll
    for (int nt = 0; nt < 4; ++nt) {
#pragma unroll
      for (int r = 0; r < 8; ++r) {
        float h = acc[nt][r] + b1a[nt];
        sH[w][(r + g * 8) * SH + nt * 16 + nloc] = (_Float16)(h > 0.f ? h : 0.f);
      }
    }
    __syncthreads();

    // hidden layer 2 (frags 4..11) then 3 (frags 12..19)
#pragma unroll
    for (int L = 0; L < 2; ++L) {
      v16h ak0 = load_a(&sH[w][0], SH, lane);
      v16h ak1 = load_a(&sH[w][32], SH, lane);
#pragma unroll
      for (int nt = 0; nt < 4; ++nt) {
        v8f c = wmma16(ak0, bf[4 + L * 8 + nt * 2 + 0], cz);
        acc[nt] = wmma16(ak1, bf[4 + L * 8 + nt * 2 + 1], c);
      }
      __syncthreads();
#pragma unroll
      for (int nt = 0; nt < 4; ++nt) {
        float b = (L == 0) ? b2a[nt] : b3a[nt];
#pragma unroll
        for (int r = 0; r < 8; ++r) {
          float h = acc[nt][r] + b;
          sH[w][(r + g * 8) * SH + nt * 16 + nloc] = (_Float16)(h > 0.f ? h : 0.f);
        }
      }
      __syncthreads();
    }

    v16h a40 = load_a(&sH[w][0], SH, lane);
    v16h a41 = load_a(&sH[w][32], SH, lane);
    v8f c4 = wmma16(a40, bf[20], cz);
    c4 = wmma16(a41, bf[21], c4);

    if (nloc == 0) {
#pragma unroll
      for (int r = 0; r < 8; ++r) {
        int n = tile * 16 + r + g * 8;
        out[n] = yp[n] + c4[r] + b4;
      }
    }
    __syncthreads();
  }
}

// ---------------------------------------------------------------------------
extern "C" void kernel_launch(void* const* d_in, const int* in_sizes, int n_in,
                              void* d_out, int out_size, void* d_ws,
                              size_t ws_size, hipStream_t stream) {
  const float* X   = (const float*)d_in[0];
  const float* yp  = (const float*)d_in[1];
  const int* edge  = (const int*)d_in[2];
  const float* ew1 = (const float*)d_in[3];  const float* eb1 = (const float*)d_in[4];
  const float* ew2 = (const float*)d_in[5];  const float* eb2 = (const float*)d_in[6];
  const float* ew3 = (const float*)d_in[7];  const float* eb3 = (const float*)d_in[8];
  const float* nw1 = (const float*)d_in[9];  const float* nb1 = (const float*)d_in[10];
  const float* nw2 = (const float*)d_in[11]; const float* nb2 = (const float*)d_in[12];
  const float* nw3 = (const float*)d_in[13]; const float* nb3 = (const float*)d_in[14];
  const float* dw1 = (const float*)d_in[15]; const float* db1 = (const float*)d_in[16];
  const float* dw2 = (const float*)d_in[17]; const float* db2 = (const float*)d_in[18];
  const float* dw3 = (const float*)d_in[19]; const float* db3 = (const float*)d_in[20];
  const float* dw4 = (const float*)d_in[21]; const float* db4 = (const float*)d_in[22];

  char* ws = (char*)d_ws;
  float* x_in   = (float*)(ws + 0);                                  // N*5 f32
  float* ea     = (float*)(ws + (size_t)1  * (1u << 20));            // E*3 f32
  float* aggr   = (float*)(ws + (size_t)11 * (1u << 20));            // N*3 f32
  float* cnt    = (float*)(ws + (size_t)12 * (1u << 20));            // N   f32
  _Float16* wf  = (_Float16*)(ws + (size_t)12 * (1u << 20) + (1u << 19));

  k_pack<<<(NFRAG * 32 + 255) / 256, 256, 0, stream>>>(
      ew1, ew2, ew3, nw1, nw2, nw3, dw1, dw2, dw3, dw4, wf);
  k_node_init<<<(NN + 255) / 256, 256, 0, stream>>>(X, yp, x_in, cnt);
  k_edge_init<<<(EE + 255) / 256, 256, 0, stream>>>(edge, x_in, ea, cnt);

  for (int it = 0; it < 3; ++it) {
    k_zero<<<(NN * 3 + 255) / 256, 256, 0, stream>>>(aggr, NN * 3);
    k_edge_mlp<<<EE / (16 * 8 * TQ_E), 256, 0, stream>>>(edge, x_in, ea, aggr,
                                                         wf, eb1, eb2, eb3);
    k_node_mlp<<<NN / (16 * 5 * TQ_N), 160, 0, stream>>>(x_in, aggr, cnt, wf,
                                                         nb1, nb2, nb3);
  }

  k_decoder<<<NN / (16 * 5 * TQ_N), 160, 0, stream>>>(x_in, yp, wf, db1, db2,
                                                      db3, db4, (float*)d_out);
}
